// Model_12266426598009
// MI455X (gfx1250) — matrix operand order repaired and checked by reading, not compiled, not run on hardware
//
#include <hip/hip_runtime.h>

// ---------------- constants ----------------
#define NBATCH   8
#define NEVENTS  8
#define BE       64
#define CDIM     128
#define NSAMP    32768
#define NFRM     128
#define NHARM    16
#define NBAND    17
#define NNOTES   128
#define NROOMS   8
#define FILTLEN  256
#define SQRT_NS  181.01933598375618f
#define PI_F     3.14159265358979323846f
#define BSTRIDE  72   // LDS B-tile row stride in f16 for 64-wide K (16B-aligned)

typedef _Float16 f16;
typedef __attribute__((ext_vector_type(16))) _Float16 v16h;
typedef __attribute__((ext_vector_type(8)))  _Float16 f16x8;
typedef __attribute__((ext_vector_type(8)))  float    v8f;
typedef __attribute__((ext_vector_type(4)))  int      v4i;

// ---------------- async global->LDS (gfx1250, ASYNCcnt) ----------------
#if defined(__gfx1250__) && \
    __has_builtin(__builtin_amdgcn_global_load_async_to_lds_b128) && \
    __has_builtin(__builtin_amdgcn_s_wait_asynccnt)
#define HAVE_ASYNC_LDS 1
#else
#define HAVE_ASYNC_LDS 0
#endif

#if HAVE_ASYNC_LDS
typedef __attribute__((address_space(1))) v4i gv4i;   // global (printed __device__)
typedef __attribute__((address_space(3))) v4i lv4i;   // LDS    (printed __shared__)
#endif

__device__ __forceinline__ void async_copy_f16x8(f16* ldst, const f16* gsrc) {
#if HAVE_ASYNC_LDS
  __builtin_amdgcn_global_load_async_to_lds_b128((gv4i*)gsrc, (lv4i*)ldst, 0, 0);
#else
  *(f16x8*)ldst = *(const f16x8*)gsrc;
#endif
}
__device__ __forceinline__ void async_join() {
#if HAVE_ASYNC_LDS
  __builtin_amdgcn_s_wait_asynccnt(0);
#endif
}

// ---------------- WMMA helpers (gfx1250, wave32) ----------------
__device__ __forceinline__ v8f wmma_f32_16x16x32(v16h a, v16h b, v8f c) {
  return __builtin_amdgcn_wmma_f32_16x16x32_f16(false, a, false, b, (short)0, c,
                                                false, false);
}
// A layout used here (16x32 f16): lane holds row m = lane&15; 16 contiguous k at
// k0 + (laneHi?16:0).
// B layout used here (32x16 f16): lane holds col n = lane&15; 16 contiguous k at
// k0 + (laneHi?16:0).   C/D: vgpr r, lane -> m = r + 8*(lane>=16), n = lane&15.

__device__ __forceinline__ v16h frag_from_2x8(f16x8 lo, f16x8 hi8) {
  v16h a;
#pragma unroll
  for (int q = 0; q < 8; ++q) { a[q] = lo[q]; a[q + 8] = hi8[q]; }
  return a;
}
__device__ __forceinline__ v16h load_a_row_f32(const float* __restrict__ row,
                                               int k0, int lane) {
  int hi = (lane & 16) ? 16 : 0;
  const float4* p = (const float4*)(row + k0 + hi);
  const float4* q = (const float4*)(row + k0 + hi + 8);
  float4 a0 = p[0], a1 = p[1], b0 = q[0], b1 = q[1];
  v16h a;
  a[0] = (f16)a0.x; a[1] = (f16)a0.y; a[2]  = (f16)a0.z; a[3]  = (f16)a0.w;
  a[4] = (f16)a1.x; a[5] = (f16)a1.y; a[6]  = (f16)a1.z; a[7]  = (f16)a1.w;
  a[8] = (f16)b0.x; a[9] = (f16)b0.y; a[10] = (f16)b0.z; a[11] = (f16)b0.w;
  a[12] = (f16)b1.x; a[13] = (f16)b1.y; a[14] = (f16)b1.z; a[15] = (f16)b1.w;
  return a;
}
__device__ __forceinline__ v16h load_a_row_f16(const f16* __restrict__ row,
                                               int k0, int lane) {
  int hi = (lane & 16) ? 16 : 0;
  f16x8 lo = *(const f16x8*)(row + k0 + hi);
  f16x8 hh = *(const f16x8*)(row + k0 + hi + 8);
  return frag_from_2x8(lo, hh);
}
__device__ __forceinline__ v16h load_16_aligned(const f16* p) {
  f16x8 lo = *(const f16x8*)p;
  f16x8 hh = *(const f16x8*)(p + 8);
  return frag_from_2x8(lo, hh);
}
__device__ __forceinline__ v16h lds_frag_contig(const f16* buf, int base) {
  v16h r;
#pragma unroll
  for (int q = 0; q < 16; ++q) r[q] = buf[base + q];
  return r;
}
__device__ __forceinline__ v16h lds_frag_a(const f16* buf, int base) {
  v16h r;
#pragma unroll
  for (int q = 0; q < 8; ++q) { r[q] = buf[base + q]; r[q + 8] = buf[base + 8 + q]; }
  return r;
}

__device__ __forceinline__ float apply_act(float v, int act) {
  switch (act) {
    case 1: return v > 0.f ? v : 0.2f * v;
    case 2: return 1.f / (1.f + __expf(-v));
    case 3: return tanhf(v);
    default: return v;
  }
}
__device__ __forceinline__ int iclamp(int v, int lo, int hi) {
  return v < lo ? lo : (v > hi ? hi : v);
}

// ========== generic fused GEMM: Y[M,N] = act(X[M,K] @ W[K,N] + b) ==========
// K % 64 == 0. Padded rows/cols via clamped indices (their C is not written).
__global__ void __launch_bounds__(256)
k_gemm(const float* __restrict__ X, const float* __restrict__ W,
       const float* __restrict__ bias, float* __restrict__ Y,
       int M, int N, int K, int act) {
  __shared__ __attribute__((aligned(16))) f16 s_b[128 * BSTRIDE];
  int lane = threadIdx.x & 31, wave = threadIdx.x >> 5;
  int nb0 = blockIdx.x * 128;
  int m0 = blockIdx.y * 16;
  int mrow = m0 + (lane & 15); if (mrow >= M) mrow = M - 1;
  const float* xrow = X + (size_t)mrow * K;
  int brow = (wave * 16 + (lane & 15)) * BSTRIDE + ((lane & 16) ? 16 : 0);
  v8f acc0 = {}, acc1 = {};
  for (int k0 = 0; k0 < K; k0 += 64) {
    __syncthreads();
    for (int li = threadIdx.x; li < 64 * 128; li += 256) {
      int kk = li >> 7, j = li & 127;
      int nn = nb0 + j; if (nn >= N) nn = N - 1;
      s_b[j * BSTRIDE + kk] = (f16)W[(size_t)(k0 + kk) * N + nn];
    }
    __syncthreads();
    v16h a0 = load_a_row_f32(xrow, k0, lane);
    v16h b0 = load_16_aligned(s_b + brow);
    acc0 = wmma_f32_16x16x32(a0, b0, acc0);
    v16h a1 = load_a_row_f32(xrow, k0 + 32, lane);
    v16h b1 = load_16_aligned(s_b + brow + 32);
    acc1 = wmma_f32_16x16x32(a1, b1, acc1);
  }
  int n0 = nb0 + wave * 16;
#pragma unroll
  for (int r = 0; r < 8; ++r) {
    int mm = m0 + r + ((lane & 16) ? 8 : 0);
    int nn = n0 + (lane & 15);
    if (mm < M && nn < N)
      Y[(size_t)mm * N + nn] = apply_act(acc0[r] + acc1[r] + bias[nn], act);
  }
}

// ====== ConvUpsample stage: repeat2 + conv(K=3,'SAME') + lrelu as im2col GEMM
__global__ void __launch_bounds__(256)
k_conv_gemm(const float* __restrict__ hin, const float* __restrict__ Wt,
            const float* __restrict__ bias, float* __restrict__ hout, int Lin) {
  __shared__ __attribute__((aligned(16))) f16 s_b[128 * BSTRIDE];
  const int Lout = Lin * 2;
  int lane = threadIdx.x & 31, wave = threadIdx.x >> 5;
  int nb0 = blockIdx.x * 128;          // Ncols = 64*Lout is a multiple of 128
  int co0 = blockIdx.y * 16;
  const float* arow = Wt + (size_t)(co0 + (lane & 15)) * 384;
  int brow = (wave * 16 + (lane & 15)) * BSTRIDE + ((lane & 16) ? 16 : 0);
  v8f acc0 = {}, acc1 = {};
  for (int k0 = 0; k0 < 384; k0 += 64) {
    __syncthreads();
    for (int li = threadIdx.x; li < 64 * 128; li += 256) {
      int kk = li >> 7, j = li & 127;
      int jg = nb0 + j;
      int ev = jg / Lout, l = jg - ev * Lout;
      int kb = k0 + kk, ci = kb / 3, t = kb - ci * 3;
      int ls = l + t - 1;
      float bv = ((unsigned)ls < (unsigned)Lout)
                     ? hin[((size_t)ev * CDIM + ci) * Lin + (ls >> 1)] : 0.f;
      s_b[j * BSTRIDE + kk] = (f16)bv;
    }
    __syncthreads();
    v16h a0 = load_a_row_f32(arow, k0, lane);
    v16h b0 = load_16_aligned(s_b + brow);
    acc0 = wmma_f32_16x16x32(a0, b0, acc0);
    v16h a1 = load_a_row_f32(arow, k0 + 32, lane);
    v16h b1 = load_16_aligned(s_b + brow + 32);
    acc1 = wmma_f32_16x16x32(a1, b1, acc1);
  }
#pragma unroll
  for (int r = 0; r < 8; ++r) {
    int co = co0 + r + ((lane & 16) ? 8 : 0);
    int j = nb0 + wave * 16 + (lane & 15);
    int ev = j / Lout, l = j - ev * Lout;
    float v = acc0[r] + acc1[r] + bias[co];
    hout[((size_t)ev * CDIM + co) * Lout + l] = v > 0.f ? v : 0.2f * v;
  }
}

// final 128->1 conv (K=3) + activation; hin [64,128,128] -> out [64,128]
__global__ void k_conv_final(const float* __restrict__ hin,
                             const float* __restrict__ ow,
                             const float* __restrict__ ob,
                             float* __restrict__ out, int act) {
  int idx = blockIdx.x * blockDim.x + threadIdx.x;
  if (idx >= BE * NFRM) return;
  int ev = idx >> 7, l = idx & 127;
  float s = ob[0];
  for (int ci = 0; ci < CDIM; ++ci) {
    const float* hp = hin + ((size_t)ev * CDIM + ci) * NFRM;
#pragma unroll
    for (int t = 0; t < 3; ++t) {
      int ls = l + t - 1;
      if ((unsigned)ls < (unsigned)NFRM) s += hp[ls] * ow[ci * 3 + t];
    }
  }
  out[idx] = apply_act(s, act);
}

// ================= small kernels =================
__global__ void k_g_max(const float* __restrict__ x, float* __restrict__ g) {
  int i = blockIdx.x * blockDim.x + threadIdx.x;
  if (i >= NBATCH * CDIM) return;
  int b = i >> 7, c = i & 127;
  float m = -1e30f;
  for (int e = 0; e < NEVENTS; ++e) m = fmaxf(m, x[(b * NEVENTS + e) * CDIM + c]);
  g[i] = m;
}

__global__ void k_gumbel_f0(const float* __restrict__ logits,
                            const float* __restrict__ gno,
                            float* __restrict__ f0_val) {
  __shared__ float sv[128]; __shared__ int si[128];
  int e = blockIdx.x, t = threadIdx.x;
  sv[t] = logits[e * NNOTES + t] + gno[e * NNOTES + t];
  si[t] = t;
  __syncthreads();
  for (int s = 64; s > 0; s >>= 1) {
    if (t < s && sv[t + s] > sv[t]) { sv[t] = sv[t + s]; si[t] = si[t + s]; }
    __syncthreads();
  }
  if (t == 0) {
    float cf = (20.f / 11025.f) * __expf(si[0] * (__logf(496.125f) / 127.f));
    f0_val[e] = cf;   // geomspace(20, 11025*0.9, 128)/11025
  }
}

__global__ void k_room_select(const float* __restrict__ h,
                              const float* __restrict__ ow,
                              const float* __restrict__ ob,
                              const float* __restrict__ gno,
                              int* __restrict__ room_idx) {
  int b = threadIdx.x;
  if (b >= NBATCH) return;
  float best = -1e30f; int bi = 0;
  for (int r = 0; r < NROOMS; ++r) {
    float s = ob[r] + gno[b * NROOMS + r];
    for (int c = 0; c < CDIM; ++c) s += h[b * CDIM + c] * ow[c * NROOMS + r];
    if (s > best) { best = s; bi = r; }
  }
  room_idx[b] = bi;
}

__global__ void k_mx(const float* __restrict__ h, const float* __restrict__ ow,
                     const float* __restrict__ ob, float* __restrict__ mx) {
  int b = threadIdx.x;
  if (b >= NBATCH) return;
  float s = ob[0];
  for (int c = 0; c < CDIM; ++c) s += h[b * CDIM + c] * ow[c];
  mx[b] = 1.f / (1.f + __expf(-s));
}

__global__ void k_make_frames(const float* __restrict__ f0_val,
                              const float* __restrict__ f0ch,
                              const float* __restrict__ ampF,
                              const float* __restrict__ decF,
                              const float* __restrict__ ss,
                              float* __restrict__ tonesF, float* __restrict__ awdF) {
  int i = blockIdx.x * blockDim.x + threadIdx.x;
  if (i >= BE * NBAND * NFRM) return;
  int fr = i & 127, band = (i >> 7) % NBAND, e = i / (NBAND * NFRM);
  float freq = f0_val[e] * (1.f + 0.1f * f0ch[e * NFRM + fr]);
  float hmul = (band == 0) ? 1.f : (1.f + (band - 1) * (16.f / 15.f));
  float tone = freq * hmul;
  tonesF[i] = (tone >= 1.f) ? 0.f : tone;
  float fscale = (band == 0) ? 1.f : ss[e * NHARM + (band - 1)];
  float af = fscale * ampF[e * NFRM + fr];
  float awd = af;
  if (fr > 0) awd = af + fscale * ampF[e * NFRM + fr - 1] * decF[e * NFRM + fr];
  awdF[i] = awd;
}

__device__ __forceinline__ float interp128(const float* f, int n) {
  float src = (n + 0.5f) * (128.f / 32768.f) - 0.5f;
  float fs = floorf(src);
  int i0 = (int)fs;
  float fr = src - fs;
  int ia = iclamp(i0, 0, 127), ib = iclamp(i0 + 1, 0, 127);
  return f[ia] + (f[ib] - f[ia]) * fr;
}

// osc = sin(cumsum(resize(tones)*pi)); accumulate sum_b osc*awd into sumOsc,
// stash first 256 osc samples (noise-filter taps) as f16.
__global__ void __launch_bounds__(256)
k_osc(const float* __restrict__ tonesF, const float* __restrict__ awdF,
      float* __restrict__ sumOsc, f16* __restrict__ filt) {
  int e = blockIdx.x, t = threadIdx.x;
  __shared__ float s_sc[256];
  __shared__ float s_carry;
  for (int band = 0; band < NBAND; ++band) {
    const float* tf = tonesF + (e * NBAND + band) * NFRM;
    const float* af = awdF + (e * NBAND + band) * NFRM;
    if (t == 0) s_carry = 0.f;
    __syncthreads();
    for (int chunk = 0; chunk < NSAMP; chunk += 4096) {
      int base = chunk + t * 16;
      float vals[16];
      float run = 0.f;
#pragma unroll
      for (int u = 0; u < 16; ++u) {
        run += interp128(tf, base + u) * PI_F;
        vals[u] = run;
      }
      s_sc[t] = run;
      __syncthreads();
      for (int off = 1; off < 256; off <<= 1) {
        float add = (t >= off) ? s_sc[t - off] : 0.f;
        __syncthreads();
        s_sc[t] += add;
        __syncthreads();
      }
      float prev = (t == 0) ? 0.f : s_sc[t - 1];
      float carry = s_carry;
      __syncthreads();
#pragma unroll
      for (int u = 0; u < 16; ++u) {
        int n = base + u;
        float osc = __sinf(carry + prev + vals[u]);
        float part = osc * interp128(af, n);
        if (band == 0) sumOsc[(size_t)e * NSAMP + n] = part;
        else           sumOsc[(size_t)e * NSAMP + n] += part;
        if (n < FILTLEN) filt[(e * NBAND + band) * FILTLEN + n] = (f16)osc;
      }
      if (t == 255) s_carry = carry + s_sc[255];
      __syncthreads();
    }
  }
}

// ===== band-limited noise via Toeplitz WMMA (ortho FFT conv == circconv/sqrtN)
// + fused mix epilogue: events = mix*sumOsc + (1-mix)*sum_b blnoise_b
__global__ void __launch_bounds__(256)
k_blnoise(const f16* __restrict__ filt, const float* __restrict__ noise,
          const float* __restrict__ mixF, float* __restrict__ events,
          f16* __restrict__ events16) {
  int e = blockIdx.y;
  int n0 = blockIdx.x * 128;
  int lane = threadIdx.x & 31, wave = threadIdx.x >> 5;
  __shared__ __attribute__((aligned(16))) f16 s_nz[384];   // reversed window
  __shared__ __attribute__((aligned(16))) f16 s_f[NBAND * FILTLEN];
  const float* np_ = noise + (size_t)e * NSAMP;
  const f16* fb = filt + (size_t)e * NBAND * FILTLEN;
  for (int li = threadIdx.x; li < (NBAND * FILTLEN) / 8; li += 256)
    async_copy_f16x8(s_f + li * 8, fb + li * 8);
  for (int i = threadIdx.x; i < 384; i += 256)
    s_nz[i] = (f16)np_[(n0 + 127 - i) & (NSAMP - 1)];
  async_join();
  __syncthreads();
  int ncol = lane & 15;
  int n = n0 + wave * 16 + ncol;
  const f16* arow0 = s_f + (lane & 15) * FILTLEN;   // bands 0..15
  const f16* arow1 = s_f + 16 * FILTLEN;            // band 16
  bool row1_valid = (lane & 15) == 0;               // rows 17..31 must be 0
  v16h zero16 = {};
  int bbase0 = (n0 + 127 - n) + ((lane & 16) ? 16 : 0);
  v8f acc0 = {}, acc1 = {};
  for (int k0 = 0; k0 < FILTLEN; k0 += 32) {
    v16h a0 = load_a_row_f16(arow0, k0, lane);
    v16h a1 = row1_valid ? load_a_row_f16(arow1, k0, lane) : zero16;
    v16h b = lds_frag_contig(s_nz, bbase0 + k0);    // noise[(n-k) mod NS]
    acc0 = wmma_f32_16x16x32(a0, b, acc0);
    acc1 = wmma_f32_16x16x32(a1, b, acc1);
  }
  float s = 0.f;
#pragma unroll
  for (int r = 0; r < 8; ++r) s += acc0[r] + acc1[r];  // reduce over bands
  s += __shfl_xor(s, 16, 32);                          // join row halves
  if (lane < 16) {
    float mix = interp128(mixF + e * NFRM, n);
    float outv = mix * events[(size_t)e * NSAMP + n] + (1.f - mix) * (s / SQRT_NS);
    events[(size_t)e * NSAMP + n] = outv;
    events16[(size_t)e * NSAMP + n] = (f16)outv;
  }
}

// ===== fits[b,e,m] ~ sum_k events[e,k]*orig[(m-k) mod NS] (scale irrelevant)
__global__ void __launch_bounds__(256)
k_fits(const f16* __restrict__ events16, const f16* __restrict__ orig16,
       float* __restrict__ fits) {
  int b = blockIdx.y;
  int m0 = blockIdx.x * 128;
  int lane = threadIdx.x & 31, wave = threadIdx.x >> 5;
  int m = m0 + wave * 16 + (lane & 15);
  __shared__ __attribute__((aligned(16))) f16 s_rev[2176];     // reversed orig
  __shared__ __attribute__((aligned(16))) f16 s_a[8 * 2048];   // event rows
  const f16* ob = orig16 + (size_t)b * NSAMP;
  const f16* eb = events16 + (size_t)b * NEVENTS * NSAMP;
  const f16* arow = s_a + (lane & 7) * 2048;
  int bb = (m0 + 127 - m) + ((lane & 16) ? 16 : 0);
  v8f acc0 = {}, acc1 = {};
  for (int kc = 0; kc < NSAMP; kc += 2048) {
    __syncthreads();
    // A rows shared by all 8 waves -> bulk async copy to LDS
    for (int li = threadIdx.x; li < 2048; li += 256) {
      int er = li >> 8, c = li & 255;
      async_copy_f16x8(s_a + er * 2048 + c * 8,
                       eb + (size_t)er * NSAMP + kc + c * 8);
    }
    // reversed (ascending-in-k) orig window, gathered per element
    for (int i = threadIdx.x; i < 2176; i += 256)
      s_rev[i] = ob[(m0 + 127 - kc - i) & (NSAMP - 1)];
    async_join();
    __syncthreads();
    for (int k0 = 0; k0 < 2048; k0 += 64) {   // dual acc breaks RAW chain
      v16h a0 = load_a_row_f16(arow, k0, lane);
      v16h b0 = lds_frag_contig(s_rev, bb + k0);
      acc0 = wmma_f32_16x16x32(a0, b0, acc0);
      v16h a1 = load_a_row_f16(arow, k0 + 32, lane);
      v16h b1 = lds_frag_contig(s_rev, bb + k0 + 32);
      acc1 = wmma_f32_16x16x32(a1, b1, acc1);
    }
  }
#pragma unroll
  for (int r = 0; r < 8; ++r) {
    int erow = r + ((lane & 16) ? 8 : 0);
    int mm = m0 + wave * 16 + (lane & 15);
    if (erow < 8)
      fits[((size_t)b * NEVENTS + erow) * NSAMP + mm] = acc0[r] + acc1[r];
  }
}

__global__ void k_argmax(const float* __restrict__ fits, int* __restrict__ idx) {
  int row = blockIdx.x;
  __shared__ float sv[256]; __shared__ int si[256];
  float best = -1e30f; int bi = 0;
  for (int n = threadIdx.x; n < NSAMP; n += 256) {
    float v = fits[(size_t)row * NSAMP + n];
    if (v > best) { best = v; bi = n; }
  }
  sv[threadIdx.x] = best; si[threadIdx.x] = bi;
  __syncthreads();
  for (int s = 128; s > 0; s >>= 1) {
    if (threadIdx.x < s) {
      bool take = sv[threadIdx.x + s] > sv[threadIdx.x] ||
                  (sv[threadIdx.x + s] == sv[threadIdx.x] &&
                   si[threadIdx.x + s] < si[threadIdx.x]);
      if (take) { sv[threadIdx.x] = sv[threadIdx.x + s]; si[threadIdx.x] = si[threadIdx.x + s]; }
    }
    __syncthreads();
  }
  if (threadIdx.x == 0) idx[row] = si[0];
}

__global__ void k_shift_sum(const float* __restrict__ events,
                            const int* __restrict__ idx,
                            float* __restrict__ ev, f16* __restrict__ ev16) {
  int b = blockIdx.y;
  int n = blockIdx.x * blockDim.x + threadIdx.x;
  if (n >= NSAMP) return;
  float s = 0.f;
  for (int e = 0; e < NEVENTS; ++e) {
    int srcn = n - idx[b * NEVENTS + e];
    if (srcn >= 0) s += events[((size_t)b * NEVENTS + e) * NSAMP + srcn];
  }
  ev[(size_t)b * NSAMP + n] = s;
  ev16[(size_t)b * NSAMP + n] = (f16)s;
}

__global__ void k_orig_to_f16(const float* __restrict__ src, f16* __restrict__ dst,
                              int n) {
  int i = blockIdx.x * blockDim.x + threadIdx.x;
  if (i < n) dst[i] = (f16)src[i];
}

__global__ void k_ir_gather(const float* __restrict__ rooms,
                            const int* __restrict__ room_idx,
                            f16* __restrict__ ir16) {
  int i = blockIdx.x * blockDim.x + threadIdx.x;
  if (i >= NBATCH * NSAMP) return;
  int b = i >> 15, n = i & (NSAMP - 1);
  ir16[i] = (f16)rooms[(size_t)room_idx[b] * NSAMP + n];
}

// ===== reverb via output-phase Toeplitz GEMM:
// wet[16t+i] = (1/sqrtN) * sum_k ir[(i-k) mod N] * ev[(k+16t) mod N]
// fused final mix: out = ev*mx + wet*(1-mx)
__global__ void __launch_bounds__(256)
k_reverb(const f16* __restrict__ ev16, const f16* __restrict__ ir16,
         const float* __restrict__ ev, const float* __restrict__ mx,
         float* __restrict__ out) {
  int b = blockIdx.y;
  int lane = threadIdx.x & 31, wave = threadIdx.x >> 5;
  int t0b = blockIdx.x * 128;
  int tcol = t0b + wave * 16 + (lane & 15);
  int irow = lane & 15;
  const f16* irb = ir16 + (size_t)b * NSAMP;
  const f16* evb = ev16 + (size_t)b * NSAMP;
  __shared__ __attribute__((aligned(16))) f16 s_ir[2064];   // reversed ir
  __shared__ __attribute__((aligned(16))) f16 s_ev[4096];   // ev window
  int ahi = (lane & 16) ? 16 : 0;
  int bhi = (lane & 16) ? 16 : 0;
  int abase = (15 - irow) + ahi;
  int bbase = 16 * (tcol - t0b) + bhi;
  v8f acc0 = {}, acc1 = {};
  for (int kc = 0; kc < NSAMP; kc += 2048) {
    __syncthreads();
    int evbase = (kc + 16 * t0b) & (NSAMP - 1);
    for (int li = threadIdx.x; li < 512; li += 256)   // 4096 f16 window
      async_copy_f16x8(s_ev + li * 8, evb + ((evbase + li * 8) & (NSAMP - 1)));
    for (int i = threadIdx.x; i < 2064; i += 256)
      s_ir[i] = irb[(15 - kc - i) & (NSAMP - 1)];
    async_join();
    __syncthreads();
    for (int k0 = 0; k0 < 2048; k0 += 64) {   // dual acc
      v16h a0 = lds_frag_a(s_ir, abase + k0);
      v16h b0 = lds_frag_contig(s_ev, bbase + k0);
      acc0 = wmma_f32_16x16x32(a0, b0, acc0);
      v16h a1 = lds_frag_a(s_ir, abase + k0 + 32);
      v16h b1 = lds_frag_contig(s_ev, bbase + k0 + 32);
      acc1 = wmma_f32_16x16x32(a1, b1, acc1);
    }
  }
  float mv = mx[b];
#pragma unroll
  for (int r = 0; r < 8; ++r) {
    int i = r + ((lane & 16) ? 8 : 0);
    int m = 16 * tcol + i;
    float wet = (acc0[r] + acc1[r]) / SQRT_NS;
    out[(size_t)b * NSAMP + m] = ev[(size_t)b * NSAMP + m] * mv + wet * (1.f - mv);
  }
}

// ================= host orchestration =================
extern "C" void kernel_launch(void* const* d_in, const int* in_sizes, int n_in,
                              void* d_out, int out_size, void* d_ws, size_t ws_size,
                              hipStream_t stream) {
  (void)in_sizes; (void)n_in; (void)out_size; (void)ws_size;
  const float* x_lat  = (const float*)d_in[0];
  const float* orig   = (const float*)d_in[1];
  const float* noise  = (const float*)d_in[2];
  const float* g_f0   = (const float*)d_in[3];
  const float* g_room = (const float*)d_in[4];
  const float* f0_w   = (const float*)d_in[5];
  const float* f0_b   = (const float*)d_in[6];
  const float* f0_ow  = (const float*)d_in[7];
  const float* f0_ob  = (const float*)d_in[8];
  const float* ss_w   = (const float*)d_in[9];
  const float* ss_b   = (const float*)d_in[10];
  const float* ss_ow  = (const float*)d_in[11];
  const float* ss_ob  = (const float*)d_in[12];
  const float* cu_pw  = (const float*)d_in[13];
  const float* cu_pb  = (const float*)d_in[14];
  const float* cu_cw  = (const float*)d_in[15];
  const float* cu_cb  = (const float*)d_in[16];
  const float* cu_ow  = (const float*)d_in[17];
  const float* cu_ob  = (const float*)d_in[18];
  const float* tm_w   = (const float*)d_in[19];
  const float* tm_b   = (const float*)d_in[20];
  const float* tm_ow  = (const float*)d_in[21];
  const float* tm_ob  = (const float*)d_in[22];
  const float* tr_w   = (const float*)d_in[23];
  const float* tr_b   = (const float*)d_in[24];
  const float* tr_ow  = (const float*)d_in[25];
  const float* tr_ob  = (const float*)d_in[26];
  const float* rooms  = (const float*)d_in[27];

  char* wsp = (char*)d_ws;
  size_t off = 0;
  auto alloc = [&](size_t bytes) -> void* {
    void* p = wsp + off;
    off += (bytes + 255) & ~(size_t)255;
    return p;
  };
  float* g      = (float*)alloc(NBATCH * CDIM * 4);
  float* gA     = (float*)alloc(NBATCH * CDIM * 4);
  float* gB     = (float*)alloc(NBATCH * CDIM * 4);
  float* losA   = (float*)alloc(BE * CDIM * 8 * 4);
  float* losB   = (float*)alloc(BE * CDIM * 8 * 4);
  float* hP     = (float*)alloc(BE * CDIM * NFRM * 4);
  float* hQ     = (float*)alloc(BE * CDIM * NFRM * 4);
  float* f0_val = (float*)alloc(BE * 4);
  float* ss_sig = (float*)alloc(BE * NHARM * 4);
  float* f0ch   = (float*)alloc(BE * NFRM * 4);
  float* ampF   = (float*)alloc(BE * NFRM * 4);
  float* decF   = (float*)alloc(BE * NFRM * 4);
  float* mixF   = (float*)alloc(BE * NFRM * 4);
  float* tonesF = (float*)alloc(BE * NBAND * NFRM * 4);
  float* awdF   = (float*)alloc(BE * NBAND * NFRM * 4);
  int*   room_i = (int*)  alloc(NBATCH * 4);
  float* mxv    = (float*)alloc(NBATCH * 4);
  float* events = (float*)alloc((size_t)BE * NSAMP * 4);
  f16*   filt16 = (f16*)  alloc((size_t)BE * NBAND * FILTLEN * 2);
  f16*   evt16  = (f16*)  alloc((size_t)BE * NSAMP * 2);
  f16*   orig16 = (f16*)  alloc((size_t)NBATCH * NSAMP * 2);
  float* fits   = (float*)alloc((size_t)BE * NSAMP * 4);
  int*   idxb   = (int*)  alloc(BE * 4);
  float* evbuf  = (float*)alloc((size_t)NBATCH * NSAMP * 4);
  f16*   ev16   = (f16*)  alloc((size_t)NBATCH * NSAMP * 2);
  f16*   ir16   = (f16*)  alloc((size_t)NBATCH * NSAMP * 2);

  auto gemm = [&](const float* X, const float* W, const float* bias, float* Y,
                  int M, int N, int K, int act) {
    dim3 grid((N + 127) / 128, (M + 15) / 16);
    k_gemm<<<grid, 256, 0, stream>>>(X, W, bias, Y, M, N, K, act);
  };

  // ---- f0 LinearOutputStack + gumbel-hard select
  gemm(x_lat, f0_w + 0 * CDIM * CDIM, f0_b + 0 * CDIM, losA, BE, CDIM, CDIM, 1);
  gemm(losA,  f0_w + 1 * CDIM * CDIM, f0_b + 1 * CDIM, losB, BE, CDIM, CDIM, 1);
  gemm(losB,  f0_w + 2 * CDIM * CDIM, f0_b + 2 * CDIM, losA, BE, CDIM, CDIM, 1);
  gemm(losA,  f0_ow, f0_ob, losB, BE, NNOTES, CDIM, 0);
  k_gumbel_f0<<<BE, 128, 0, stream>>>(losB, g_f0, f0_val);

  // ---- spec_shape stack
  gemm(x_lat, ss_w + 0 * CDIM * CDIM, ss_b + 0 * CDIM, losA, BE, CDIM, CDIM, 1);
  gemm(losA,  ss_w + 1 * CDIM * CDIM, ss_b + 1 * CDIM, losB, BE, CDIM, CDIM, 1);
  gemm(losB,  ss_w + 2 * CDIM * CDIM, ss_b + 2 * CDIM, losA, BE, CDIM, CDIM, 1);
  gemm(losA,  ss_ow, ss_ob, ss_sig, BE, NHARM, CDIM, 2);

  // ---- global latent: room + dry/wet mix
  k_g_max<<<(NBATCH * CDIM + 255) / 256, 256, 0, stream>>>(x_lat, g);
  gemm(g,  tr_w + 0 * CDIM * CDIM, tr_b + 0 * CDIM, gA, NBATCH, CDIM, CDIM, 1);
  gemm(gA, tr_w + 1 * CDIM * CDIM, tr_b + 1 * CDIM, gB, NBATCH, CDIM, CDIM, 1);
  k_room_select<<<1, NBATCH, 0, stream>>>(gB, tr_ow, tr_ob, g_room, room_i);
  gemm(g,  tm_w + 0 * CDIM * CDIM, tm_b + 0 * CDIM, gA, NBATCH, CDIM, CDIM, 1);
  gemm(gA, tm_w + 1 * CDIM * CDIM, tm_b + 1 * CDIM, gB, NBATCH, CDIM, CDIM, 1);
  k_mx<<<1, NBATCH, 0, stream>>>(gB, tm_ow, tm_ob, mxv);

  // ---- 4 ConvUpsamples: 0=f0_change(tanh), 1=amp(sig), 2=decay(sig), 3=mix(sig)
  float* cu_target[4] = {f0ch, ampF, decF, mixF};
  int cu_act[4] = {3, 2, 2, 2};
  for (int u = 0; u < 4; ++u) {
    gemm(x_lat, cu_pw + (size_t)u * CDIM * CDIM * 8, cu_pb + u * CDIM * 8,
         losA, BE, CDIM * 8, CDIM, 1);
    const float* cw = cu_cw + (size_t)u * 4 * CDIM * CDIM * 3;
    const float* cb = cu_cb + (size_t)u * 4 * CDIM;
    int Lin = 8;
    const float* hin = losA;
    float* houts[4] = {hP, hQ, hP, hQ};
    for (int s = 0; s < 4; ++s) {
      int N = BE * Lin * 2;
      dim3 grid((N + 127) / 128, CDIM / 16);
      k_conv_gemm<<<grid, 256, 0, stream>>>(hin, cw + (size_t)s * CDIM * CDIM * 3,
                                            cb + s * CDIM, houts[s], Lin);
      hin = houts[s];
      Lin *= 2;
    }
    k_conv_final<<<(BE * NFRM + 255) / 256, 256, 0, stream>>>(
        hQ, cu_ow + (size_t)u * CDIM * 3, cu_ob + u, cu_target[u], cu_act[u]);
  }

  // ---- oscillator bank (fused resize+cumsum+sin+awd accumulate)
  k_make_frames<<<(BE * NBAND * NFRM + 255) / 256, 256, 0, stream>>>(
      f0_val, f0ch, ampF, decF, ss_sig, tonesF, awdF);
  k_osc<<<BE, 256, 0, stream>>>(tonesF, awdF, events, filt16);

  // ---- band-limited noise (Toeplitz WMMA) + mix epilogue
  k_blnoise<<<dim3(NSAMP / 128, BE), 256, 0, stream>>>(filt16, noise, mixF,
                                                       events, evt16);

  // ---- cross-correlation (Toeplitz WMMA), argmax, shift+sum
  k_orig_to_f16<<<(NBATCH * NSAMP + 255) / 256, 256, 0, stream>>>(
      orig, orig16, NBATCH * NSAMP);
  k_fits<<<dim3(NSAMP / 128, NBATCH), 256, 0, stream>>>(evt16, orig16, fits);
  k_argmax<<<BE, 256, 0, stream>>>(fits, idxb);
  k_shift_sum<<<dim3(NSAMP / 256, NBATCH), 256, 0, stream>>>(events, idxb,
                                                             evbuf, ev16);

  // ---- neural reverb (Toeplitz WMMA) + final dry/wet mix -> d_out
  k_ir_gather<<<(NBATCH * NSAMP + 255) / 256, 256, 0, stream>>>(rooms, room_i, ir16);
  k_reverb<<<dim3((NSAMP / 16) / 128, NBATCH), 256, 0, stream>>>(
      ev16, ir16, evbuf, mxv, (float*)d_out);
}
